// Hausdorff_51969104281658
// MI455X (gfx1250) — matrix operand order repaired and checked by reading, not compiled, not run on hardware
//
#include <hip/hip_runtime.h>
#include <math.h>

#define BB 8
#define LL 2048
#define KK 2048
#define DD 64
#define NITER 100
#define LOG2E_F 1.4426950408889634f
#define LN2_F   0.6931471805599453f

typedef __attribute__((ext_vector_type(16))) _Float16 v16h;
typedef __attribute__((ext_vector_type(8)))  float    v8f;

// ---------------------------------------------------------------------------
// A-fragment (16x32 f16, MxK): lane l: M = l&15, g = l>>4.
// element e<8  -> K = kb + g*8 + e
// element e>=8 -> K = kb + 16 + g*8 + (e-8)
// Split-f16: hi = f16(x), lo = f16(x - hi)  (error compensation)
// ---------------------------------------------------------------------------
__device__ __forceinline__ void load_A_split(const float* __restrict__ row, int g,
                                             int kb, v16h& hi, v16h& lo) {
  const float* p0 = row + kb + g * 8;
  const float* p1 = p0 + 16;
#pragma unroll
  for (int e = 0; e < 8; ++e) {
    float x0 = p0[e];
    _Float16 h0 = (_Float16)x0;
    hi[e] = h0;
    lo[e] = (_Float16)(x0 - (float)h0);
    float x1 = p1[e];
    _Float16 h1 = (_Float16)x1;
    hi[e + 8] = h1;
    lo[e + 8] = (_Float16)(x1 - (float)h1);
  }
}

// ---------------------------------------------------------------------------
// B-fragment (32x16 f16, KxN): lane l: N = l&15, g = l>>4.
// element e<8 -> K = kb + g*8 + e; e>=8 -> K = kb + 16 + g*8 + (e-8)
// B[k][n] = Y[n][k]  (Y row-major, D contiguous)
// ---------------------------------------------------------------------------
__device__ __forceinline__ void load_B_split(const float* __restrict__ row, int g,
                                             int kb, v16h& hi, v16h& lo) {
  const float* p = row + kb + g * 8;
#pragma unroll
  for (int e = 0; e < 16; ++e) {
    float x = p[e + (e & 8)];
    _Float16 h = (_Float16)x;
    hi[e] = h;
    lo[e] = (_Float16)(x - (float)h);
  }
}

// ---------------------------------------------------------------------------
// 0.5 * ||row||^2 for each D=64 row
// ---------------------------------------------------------------------------
__global__ __launch_bounds__(256) void row_norms(const float* __restrict__ X,
                                                 float* __restrict__ ns, int nrows) {
  int row = blockIdx.x * 256 + threadIdx.x;
  if (row >= nrows) return;
  const float4* p = (const float4*)(X + (size_t)row * DD);
  float s = 0.f;
#pragma unroll
  for (int i = 0; i < DD / 4; ++i) {
    float4 v = p[i];
    s += v.x * v.x + v.y * v.y + v.z * v.z + v.w * v.w;
  }
  ns[row] = 0.5f * s;
}

__global__ __launch_bounds__(256) void fill_zero(float* __restrict__ p, int n) {
  int i = blockIdx.x * blockDim.x + threadIdx.x;
  if (i < n) p[i] = 0.f;
}

// ---------------------------------------------------------------------------
// K2[b][i][j] = (x_i . y_j - xs_i - ys_j) * log2(e)
// One wave per 16x16 tile; D=64 in two K=32 steps; 3 WMMAs per step (split-f16).
// Rows(X)=Rows(Y)=2048 for both self and cross cases.
// ---------------------------------------------------------------------------
__global__ __launch_bounds__(128) void sinkhorn_gemm(const float* __restrict__ X,
                                                     const float* __restrict__ Y,
                                                     const float* __restrict__ xs,
                                                     const float* __restrict__ ys,
                                                     float* __restrict__ K2) {
  const int lane = threadIdx.x;
  const int tilesY = KK / 16;                 // 128 col tiles
  const int tilesPerBatch = (LL / 16) * tilesY;
  int tile = blockIdx.x * blockDim.y + threadIdx.y;
  if (tile >= BB * tilesPerBatch) return;     // wave-uniform guard (EXEC stays full)
  int b = tile / tilesPerBatch;
  int t = tile - b * tilesPerBatch;
  int ti = t / tilesY;
  int tj = t - ti * tilesY;

  const int g = lane >> 4;
  const int r = lane & 15;
  const float* arow = X + ((size_t)b * LL + (size_t)(ti * 16 + r)) * DD;
  const float* brow = Y + ((size_t)b * KK + (size_t)(tj * 16 + r)) * DD;

  v8f acc = {};
#pragma unroll
  for (int kb = 0; kb < DD; kb += 32) {
    v16h ah, al, bh, bl;
    load_A_split(arow, g, kb, ah, al);
    load_B_split(brow, g, kb, bh, bl);
    acc = __builtin_amdgcn_wmma_f32_16x16x32_f16(false, ah, false, bh, (short)0, acc, false, false);
    acc = __builtin_amdgcn_wmma_f32_16x16x32_f16(false, ah, false, bl, (short)0, acc, false, false);
    acc = __builtin_amdgcn_wmma_f32_16x16x32_f16(false, al, false, bh, (short)0, acc, false, false);
  }

  // C/D layout: VGPR q, lane group g -> row M = q + 8g, col N = lane&15
  int col = tj * 16 + r;
  float ysv = ys[(size_t)b * KK + col];
  const float* xsb = xs + (size_t)b * LL;
  float* outb = K2 + (size_t)b * LL * KK;
#pragma unroll
  for (int q = 0; q < 8; ++q) {
    int rowi = ti * 16 + g * 8 + q;
    float v = (acc[q] - xsb[rowi] - ysv) * LOG2E_F;
    outb[(size_t)rowi * KK + col] = v;
  }
}

// ---------------------------------------------------------------------------
// One Sinkhorn half-step.  One wave per row; the whole 2048-wide row lives in
// 64 VGPRs per lane (float4 b128 loads) so it is read from L2 exactly once.
// flw2[j] = (f[j]+lw[j])*log2e staged in LDS per 8-row block.
// ---------------------------------------------------------------------------
__global__ __launch_bounds__(256) void lse_step(const float* __restrict__ K2,
                                                const float* __restrict__ fin,
                                                const float* __restrict__ lw,
                                                float* __restrict__ fout) {
  __shared__ float flw[KK];
  const int lane = threadIdx.x;          // 0..31
  const int wv = threadIdx.y;            // 0..7
  const int tid = wv * 32 + lane;
  int blockRow0 = blockIdx.x * 8;        // 8 rows per block, never straddles a batch
  int b = blockRow0 / LL;
  int row0 = blockRow0 - b * LL;
  const float* fb = fin + (size_t)b * LL;
  const float* lwb = lw + (size_t)b * LL;
#pragma unroll
  for (int k = 0; k < KK / 256; ++k) {
    int j = tid + 256 * k;
    flw[j] = (fb[j] + lwb[j]) * LOG2E_F;
  }
  __syncthreads();

  int rowi = row0 + wv;
  const float* krow = K2 + ((size_t)b * LL + rowi) * KK;
  float v[64];
  float m = -3.0e38f;
#pragma unroll
  for (int c = 0; c < 16; ++c) {
    int j = c * 128 + lane * 4;
    float4 kv = *(const float4*)(krow + j);
    float a0 = kv.x + flw[j + 0];
    float a1 = kv.y + flw[j + 1];
    float a2 = kv.z + flw[j + 2];
    float a3 = kv.w + flw[j + 3];
    v[c * 4 + 0] = a0; v[c * 4 + 1] = a1; v[c * 4 + 2] = a2; v[c * 4 + 3] = a3;
    m = fmaxf(m, fmaxf(fmaxf(a0, a1), fmaxf(a2, a3)));
  }
#pragma unroll
  for (int o = 16; o > 0; o >>= 1) m = fmaxf(m, __shfl_xor(m, o));
  float s = 0.f;
#pragma unroll
  for (int i = 0; i < 64; ++i) s += __builtin_amdgcn_exp2f(v[i] - m);
#pragma unroll
  for (int o = 16; o > 0; o >>= 1) s += __shfl_xor(s, o);
  if (lane == 0) {
    float g = -(m + __builtin_amdgcn_logf(s)) * LN2_F;   // back to natural log
    fout[(size_t)b * LL + rowi] = 0.5f * (fb[rowi] + g);
  }
}

// ---------------------------------------------------------------------------
// Final pass over Kcross: g_t = -lse_j(Kcross[t,j]+f[j]+lw[j]);
// out[b] += g_t * exp(tlw[t])
// ---------------------------------------------------------------------------
__global__ __launch_bounds__(256) void lse_final(const float* __restrict__ K2c,
                                                 const float* __restrict__ fin,
                                                 const float* __restrict__ lw,
                                                 const float* __restrict__ tlw,
                                                 float* __restrict__ out) {
  __shared__ float flw[LL];
  const int lane = threadIdx.x;
  const int wv = threadIdx.y;
  const int tid = wv * 32 + lane;
  int blockRow0 = blockIdx.x * 8;
  int b = blockRow0 / KK;
  int row0 = blockRow0 - b * KK;
  const float* fb = fin + (size_t)b * LL;
  const float* lwb = lw + (size_t)b * LL;
#pragma unroll
  for (int k = 0; k < LL / 256; ++k) {
    int j = tid + 256 * k;
    flw[j] = (fb[j] + lwb[j]) * LOG2E_F;
  }
  __syncthreads();

  int rowi = row0 + wv;
  const float* krow = K2c + ((size_t)b * KK + rowi) * LL;
  float v[64];
  float m = -3.0e38f;
#pragma unroll
  for (int c = 0; c < 16; ++c) {
    int j = c * 128 + lane * 4;
    float4 kv = *(const float4*)(krow + j);
    float a0 = kv.x + flw[j + 0];
    float a1 = kv.y + flw[j + 1];
    float a2 = kv.z + flw[j + 2];
    float a3 = kv.w + flw[j + 3];
    v[c * 4 + 0] = a0; v[c * 4 + 1] = a1; v[c * 4 + 2] = a2; v[c * 4 + 3] = a3;
    m = fmaxf(m, fmaxf(fmaxf(a0, a1), fmaxf(a2, a3)));
  }
#pragma unroll
  for (int o = 16; o > 0; o >>= 1) m = fmaxf(m, __shfl_xor(m, o));
  float s = 0.f;
#pragma unroll
  for (int i = 0; i < 64; ++i) s += __builtin_amdgcn_exp2f(v[i] - m);
#pragma unroll
  for (int o = 16; o > 0; o >>= 1) s += __shfl_xor(s, o);
  if (lane == 0) {
    float g = -(m + __builtin_amdgcn_logf(s)) * LN2_F;
    float w = __builtin_amdgcn_exp2f(tlw[(size_t)b * KK + rowi] * LOG2E_F);
    atomicAdd(&out[b], g * w);
  }
}

// ---------------------------------------------------------------------------
// Workspace layout (floats): K2[BB*LL*KK] | xs[BB*LL] | ts[BB*KK] | f0 | f1
// ~128.4 MiB; Kcross overwrites K2 after the 100 iterations.
// ---------------------------------------------------------------------------
extern "C" void kernel_launch(void* const* d_in, const int* in_sizes, int n_in,
                              void* d_out, int out_size, void* d_ws, size_t ws_size,
                              hipStream_t stream) {
  const float* pos  = (const float*)d_in[0];
  const float* lw   = (const float*)d_in[1];
  const float* tpos = (const float*)d_in[2];
  const float* tlw  = (const float*)d_in[3];
  float* out = (float*)d_out;

  float* K2 = (float*)d_ws;
  float* xs = K2 + (size_t)BB * LL * KK;
  float* ts = xs + (size_t)BB * LL;
  float* f0 = ts + (size_t)BB * KK;
  float* f1 = f0 + (size_t)BB * LL;

  row_norms<<<(BB * LL + 255) / 256, 256, 0, stream>>>(pos, xs, BB * LL);
  row_norms<<<(BB * KK + 255) / 256, 256, 0, stream>>>(tpos, ts, BB * KK);
  fill_zero<<<(BB * LL + 255) / 256, 256, 0, stream>>>(f0, BB * LL);

  const int tiles = BB * (LL / 16) * (KK / 16);   // 131072 tiles, 4 waves/block
  sinkhorn_gemm<<<tiles / 4, dim3(32, 4), 0, stream>>>(pos, pos, xs, xs, K2);

  float* fin = f0;
  float* fout = f1;
  for (int it = 0; it < NITER; ++it) {
    lse_step<<<BB * LL / 8, dim3(32, 8), 0, stream>>>(K2, fin, lw, fout);
    float* t = fin; fin = fout; fout = t;
  }
  // NITER even -> final f back in f0 (== fin)

  // Kcross overwrites K2 (Kself no longer needed)
  sinkhorn_gemm<<<tiles / 4, dim3(32, 4), 0, stream>>>(tpos, pos, ts, xs, K2);

  fill_zero<<<1, 32, 0, stream>>>(out, out_size);
  lse_final<<<BB * KK / 8, dim3(32, 8), 0, stream>>>(K2, fin, lw, tlw, out);
}
